// PhpNetGraphTokensSingle_62010737820208
// MI455X (gfx1250) — matrix operand order repaired and checked by reading, not compiled, not run on hardware
//
#include <hip/hip_runtime.h>
#include <math.h>

#define N_NODES 10000
#define E_EDGES 100000
#define NBATCH  64
#define VOCABSZ 5000
#define EMBD    100
#define HDIM    2000
#define LAYERS  4

// conv pipeline lengths
#define C1L 666   // (2000-2)=1998 conv -> /3 pool
#define C2L 221   // 666-2=664 -> 221
#define C3L 73    // 221-2=219 -> 73

typedef __attribute__((ext_vector_type(16))) __bf16 v16bf;
typedef __attribute__((ext_vector_type(8)))  float  v8f;

struct alignas(16) Q16 { unsigned int x, y, z, w; };
union FragAB { v16bf v; Q16 q[2]; };

__device__ __forceinline__ unsigned short f2bf(float f) {
  unsigned int u = __float_as_uint(f);
  u += 0x7FFFu + ((u >> 16) & 1u);      // round-to-nearest-even
  return (unsigned short)(u >> 16);
}

// ---------------- elementwise helpers ----------------
__global__ void k_f32_to_bf16(const float* __restrict__ s, unsigned short* __restrict__ d, int n) {
  int i = blockIdx.x * blockDim.x + threadIdx.x;
  if (i < n) d[i] = f2bf(s[i]);
}

__global__ void k_zero_f32(float* __restrict__ p, int n) {
  int i = blockIdx.x * blockDim.x + threadIdx.x;
  if (i < n) p[i] = 0.f;
}

__global__ void k_embed(const int* __restrict__ tokens, const float* __restrict__ table,
                        float* __restrict__ h, unsigned short* __restrict__ hbf) {
  int i = blockIdx.x * blockDim.x + threadIdx.x;
  if (i >= N_NODES * HDIM) return;
  int n = i / HDIM, j = i - n * HDIM;
  float v = (j < EMBD) ? table[tokens[n] * EMBD + j] : 0.f;
  h[i] = v; hbf[i] = f2bf(v);
}

__global__ void k_scatter_add(const int* __restrict__ ei, const float* __restrict__ m,
                              float* __restrict__ agg) {
  int e = blockIdx.x;
  int src = ei[e], dst = ei[E_EDGES + e];
  const float* ms = m + (size_t)src * HDIM;
  float* ad = agg + (size_t)dst * HDIM;
  for (int j = threadIdx.x; j < HDIM; j += blockDim.x)
    atomicAdd(&ad[j], ms[j]);
}

__global__ void k_gru(const float* __restrict__ rsum, const float* __restrict__ zsum,
                      const float* __restrict__ innb, const float* __restrict__ hnb,
                      float* __restrict__ h, unsigned short* __restrict__ hbf) {
  int i = blockIdx.x * blockDim.x + threadIdx.x;
  if (i >= N_NODES * HDIM) return;
  float r  = 1.f / (1.f + __expf(-rsum[i]));
  float z  = 1.f / (1.f + __expf(-zsum[i]));
  float nn = tanhf(innb[i] + r * hnb[i]);
  float hv = (1.f - z) * nn + z * h[i];
  h[i] = hv; hbf[i] = f2bf(hv);
}

__global__ void k_relu_segmax(const float* __restrict__ h, const int* __restrict__ batch,
                              float* __restrict__ pooled) {
  int n = blockIdx.x;
  int b = batch[n];
  const float* hr = h + (size_t)n * HDIM;
  int* pr = (int*)(pooled + (size_t)b * HDIM);
  for (int j = threadIdx.x; j < HDIM; j += blockDim.x) {
    float x = fmaxf(hr[j], 0.f);                 // x >= 0 -> int order == float order
    atomicMax(&pr[j], __float_as_int(x));
  }
}

// ---------------- WMMA bf16 GEMM ----------------
// C[M,N] (f32) = A[M,K](bf16) x B (bf16) (+ bias[n]) (+ C if accumulate)
// OPB==0: B is row-major K x N (ldb = row stride)
// OPB==1: B is row-major N x K (i.e. multiply by B^T), ldb = row stride
#define BM 128
#define BN 128
#define BK 32

// guarded staging (edge blocks / K tail only)
__device__ __forceinline__ void stage_A_g(const unsigned short* __restrict__ A, int lda,
                                          unsigned short* As, int blockM, int kk,
                                          int M, int K, int tid) {
  for (int c = tid; c < (BM * BK) / 8; c += 256) {
    int row = c >> 2, kc = (c & 3) * 8;
    int gm = blockM + row;
    unsigned short* dst = &As[row * BK + kc];
    if (gm < M && (kk + kc + 8) <= K) {
      *(Q16*)dst = *(const Q16*)&A[(size_t)gm * lda + kk + kc];
    } else {
      for (int j = 0; j < 8; ++j) {
        int k = kk + kc + j;
        dst[j] = (gm < M && k < K) ? A[(size_t)gm * lda + k] : (unsigned short)0;
      }
    }
  }
}

template<int OPB>
__device__ __forceinline__ void stage_B_g(const unsigned short* __restrict__ B, int ldb,
                                          unsigned short* Bs, int blockN, int kk,
                                          int N, int K, int tid) {
  if (OPB) {
    for (int c = tid; c < (BN * BK) / 8; c += 256) {
      int nl = c >> 2, kc = (c & 3) * 8;
      int gn = blockN + nl;
      unsigned short* dst = &Bs[nl * BK + kc];
      if (gn < N && (kk + kc + 8) <= K) {
        *(Q16*)dst = *(const Q16*)&B[(size_t)gn * ldb + kk + kc];
      } else {
        for (int j = 0; j < 8; ++j) {
          int k = kk + kc + j;
          dst[j] = (gn < N && k < K) ? B[(size_t)gn * ldb + k] : (unsigned short)0;
        }
      }
    }
  } else {
    for (int c = tid; c < (BK * BN) / 8; c += 256) {
      int kl = c >> 4, nc = (c & 15) * 8;
      int gk = kk + kl;
      unsigned short tmp[8];
      if (gk < K && (blockN + nc + 8) <= N) {
        *(Q16*)tmp = *(const Q16*)&B[(size_t)gk * ldb + blockN + nc];
      } else {
        for (int j = 0; j < 8; ++j) {
          int gn = blockN + nc + j;
          tmp[j] = (gk < K && gn < N) ? B[(size_t)gk * ldb + gn] : (unsigned short)0;
        }
      }
      for (int j = 0; j < 8; ++j) Bs[(nc + j) * BK + kl] = tmp[j];
    }
  }
}

// Fragments per ISA lane layout:
//  A 16x32 bf16: lanes0-15 K={0..7,16..23}, lanes16-31 K={8..15,24..31}
//  B 32x16:      lane col = lane&15, K same as A: lanes0-15 K={0..7,16..23}, lanes16-31 K={8..15,24..31} (n-major LDS)
__device__ __forceinline__ void compute_step(const unsigned short* As, const unsigned short* Bs,
                                             v8f acc[2][4], int wrow, int wcol,
                                             int l16, int lhi) {
  FragAB fa[2], fb[4];
  for (int t = 0; t < 2; ++t) {
    int arow = wrow * 32 + t * 16 + l16;
    fa[t].q[0] = *(const Q16*)&As[arow * BK + lhi * 8];
    fa[t].q[1] = *(const Q16*)&As[arow * BK + 16 + lhi * 8];
  }
  for (int t = 0; t < 4; ++t) {
    int bn = wcol * 64 + t * 16 + l16;
    fb[t].q[0] = *(const Q16*)&Bs[bn * BK + lhi * 8];
    fb[t].q[1] = *(const Q16*)&Bs[bn * BK + 16 + lhi * 8];
  }
  for (int tm = 0; tm < 2; ++tm)
    for (int tn = 0; tn < 4; ++tn)
      acc[tm][tn] = __builtin_amdgcn_wmma_f32_16x16x32_bf16(
          false, fa[tm].v, false, fb[tn].v, (short)0, acc[tm][tn], false, false);
}

template<int OPB>
__global__ __launch_bounds__(256) void k_gemm_bf16(
    const unsigned short* __restrict__ A, int lda,
    const unsigned short* __restrict__ B, int ldb,
    float* __restrict__ C, int ldc,
    const float* __restrict__ bias, int accumulate,
    int M, int N, int K)
{
  // ping-pong LDS: 2 x (8KB A + 8KB B) = 32KB of the 320KB WGP LDS
  __shared__ __align__(16) unsigned short As[2][BM * BK];   // row-major, stride BK
  __shared__ __align__(16) unsigned short Bs[2][BN * BK];   // n-major (transposed), stride BK

  const int tid    = threadIdx.x;
  const int lane   = tid & 31;
  const int wave   = tid >> 5;           // 0..7
  const int wrow   = wave >> 1;          // 0..3 : 32-row strip
  const int wcol   = wave & 1;           // 0..1 : 64-col strip
  const int blockN = blockIdx.x * BN;
  const int blockM = blockIdx.y * BM;
  const int l16    = lane & 15;
  const int lhi    = lane >> 4;

  v8f acc[2][4];
  const v8f vzero = {0.f, 0.f, 0.f, 0.f, 0.f, 0.f, 0.f, 0.f};
  for (int tm = 0; tm < 2; ++tm)
    for (int tn = 0; tn < 4; ++tn) acc[tm][tn] = vzero;

  const int ksteps    = (K + BK - 1) / BK;
  const int kfull     = K / BK;
  const bool interior = (blockM + BM <= M) && (blockN + BN <= N);

  int ks = 0;
  if (interior && kfull > 0) {
    // ---- per-thread staging coordinates, hoisted out of the K loop ----
    const int arow = tid >> 2;          // 0..63 (rows arow and arow+64)
    const int akc  = (tid & 3) * 8;     // 0,8,16,24
    const int aoff = arow * BK + akc;
    const unsigned short* pA0 = A + (size_t)(blockM + arow) * lda + akc;
    const unsigned short* pA1 = pA0 + (size_t)64 * lda;

    const unsigned short* pB0;
    const unsigned short* pB1;
    int boff = 0, bkl = 0, bnc = 0;
    if (OPB) {
      boff = arow * BK + akc;
      pB0  = B + (size_t)(blockN + arow) * ldb + akc;   // rows n, n+64 of B^T
      pB1  = pB0 + (size_t)64 * ldb;
    } else {
      bkl = tid >> 4;                    // k rows bkl and bkl+16
      bnc = (tid & 15) * 8;              // 0..120
      pB0 = B + (size_t)bkl * ldb + blockN + bnc;
      pB1 = pB0 + (size_t)16 * ldb;
    }

    auto fetch = [&](Q16& a0, Q16& a1, Q16& b0, Q16& b1) {
      a0 = *(const Q16*)pA0;
      a1 = *(const Q16*)pA1;
      b0 = *(const Q16*)pB0;
      b1 = *(const Q16*)pB1;
      pA0 += BK; pA1 += BK;
      if (OPB) { pB0 += BK; pB1 += BK; }
      else     { pB0 += (size_t)32 * ldb; pB1 += (size_t)32 * ldb; }
    };
    auto stash = [&](int buf, const Q16& a0, const Q16& a1, const Q16& b0, const Q16& b1) {
      *(Q16*)&As[buf][aoff]           = a0;
      *(Q16*)&As[buf][aoff + 64 * BK] = a1;
      if (OPB) {
        *(Q16*)&Bs[buf][boff]           = b0;
        *(Q16*)&Bs[buf][boff + 64 * BK] = b1;
      } else {
        const unsigned short* t0 = (const unsigned short*)&b0;
        const unsigned short* t1 = (const unsigned short*)&b1;
        for (int j = 0; j < 8; ++j) {
          Bs[buf][(bnc + j) * BK + bkl]      = t0[j];
          Bs[buf][(bnc + j) * BK + bkl + 16] = t1[j];
        }
      }
    };

    // prologue: stage buffer 0
    {
      Q16 a0, a1, b0, b1;
      fetch(a0, a1, b0, b1);
      stash(0, a0, a1, b0, b1);
    }
    __syncthreads();

    // steady state: load next slab -> compute current -> stash next -> barrier
    for (; ks < kfull - 1; ++ks) {
      const int cur = ks & 1;
      Q16 a0, a1, b0, b1;
      fetch(a0, a1, b0, b1);
      __builtin_prefetch(pA0, 0, 1);     // next A K-slab -> global_prefetch_b8
      compute_step(As[cur], Bs[cur], acc, wrow, wcol, l16, lhi);
      stash(cur ^ 1, a0, a1, b0, b1);
      __syncthreads();
    }
    // epilogue: compute last staged buffer
    compute_step(As[(kfull - 1) & 1], Bs[(kfull - 1) & 1], acc, wrow, wcol, l16, lhi);
    __syncthreads();
    ks = kfull;
  }
  // ---- guarded path (edge blocks and K remainder), buffer 0, 2 barriers ----
  for (; ks < ksteps; ++ks) {
    const int kk = ks * BK;
    stage_A_g(A, lda, As[0], blockM, kk, M, K, tid);
    stage_B_g<OPB>(B, ldb, Bs[0], blockN, kk, N, K, tid);
    __syncthreads();
    compute_step(As[0], Bs[0], acc, wrow, wcol, l16, lhi);
    __syncthreads();
  }

  // ---- store: C/D layout: VGPR r -> M = r + 8*(lane>=16), N = lane&15 ----
  if (interior) {
    for (int tm = 0; tm < 2; ++tm) {
      for (int tn = 0; tn < 4; ++tn) {
        int gn = blockN + wcol * 64 + tn * 16 + l16;
        int mb = blockM + wrow * 32 + tm * 16 + 8 * lhi;
        float bv = bias ? bias[gn] : 0.f;
        for (int r = 0; r < 8; ++r) {
          size_t o = (size_t)(mb + r) * ldc + gn;
          float v = acc[tm][tn][r] + bv;
          if (accumulate) v += C[o];
          C[o] = v;
        }
      }
    }
  } else {
    for (int tm = 0; tm < 2; ++tm) {
      for (int tn = 0; tn < 4; ++tn) {
        int gn = blockN + wcol * 64 + tn * 16 + l16;
        int mb = blockM + wrow * 32 + tm * 16 + 8 * lhi;
        if (gn >= N) continue;
        float bv = bias ? bias[gn] : 0.f;
        for (int r = 0; r < 8; ++r) {
          int gm = mb + r;
          if (gm >= M) continue;
          size_t o = (size_t)gm * ldc + gn;
          float v = acc[tm][tn][r] + bv;
          if (accumulate) v += C[o];
          C[o] = v;
        }
      }
    }
  }
}

// ---------------- conv / linear head (tiny, VALU) ----------------
__global__ void k_conv1(const float* __restrict__ in, const float* __restrict__ w,
                        const float* __restrict__ b, float* __restrict__ out) {
  int i = blockIdx.x * blockDim.x + threadIdx.x;
  if (i >= NBATCH * 50 * C1L) return;
  int xp = i % C1L; int t = i / C1L; int oc = t % 50; int bb = t / 50;
  const float* row = in + (size_t)bb * HDIM;
  float w0 = w[oc * 3], w1 = w[oc * 3 + 1], w2 = w[oc * 3 + 2], bv = b[oc];
  float mv = -1e30f;
  for (int p = 0; p < 3; ++p) {
    int x0 = xp * 3 + p;
    float v = row[x0] * w0 + row[x0 + 1] * w1 + row[x0 + 2] * w2 + bv;
    mv = fmaxf(mv, v);
  }
  out[i] = fmaxf(mv, 0.f);
}

__global__ void k_conv2(const float* __restrict__ in, const float* __restrict__ w,
                        const float* __restrict__ b, float* __restrict__ out) {
  int i = blockIdx.x * blockDim.x + threadIdx.x;
  if (i >= NBATCH * 100 * C2L) return;
  int xp = i % C2L; int t = i / C2L; int oc = t % 100; int bb = t / 100;
  float mv = -1e30f;
  for (int p = 0; p < 3; ++p) {
    int x0 = xp * 3 + p;
    float s = b[oc];
    for (int ic = 0; ic < 50; ++ic) {
      const float* r  = in + ((size_t)bb * 50 + ic) * C1L + x0;
      const float* wp = w + ((size_t)oc * 50 + ic) * 3;
      s += r[0] * wp[0] + r[1] * wp[1] + r[2] * wp[2];
    }
    mv = fmaxf(mv, s);
  }
  out[i] = fmaxf(mv, 0.f);
}

__global__ void k_conv3(const float* __restrict__ in, const float* __restrict__ w,
                        const float* __restrict__ b, float* __restrict__ out) {
  int i = blockIdx.x * blockDim.x + threadIdx.x;
  if (i >= NBATCH * 150 * C3L) return;
  int xp = i % C3L; int t = i / C3L; int oc = t % 150; int bb = t / 150;
  float mv = -1e30f;
  for (int p = 0; p < 3; ++p) {
    int x0 = xp * 3 + p;
    float s = b[oc];
    for (int ic = 0; ic < 100; ++ic) {
      const float* r  = in + ((size_t)bb * 100 + ic) * C2L + x0;
      const float* wp = w + ((size_t)oc * 100 + ic) * 3;
      s += r[0] * wp[0] + r[1] * wp[1] + r[2] * wp[2];
    }
    mv = fmaxf(mv, s);
  }
  out[i] = fmaxf(mv, 0.f);
}

__global__ void k_lin1(const float* __restrict__ x, const float* __restrict__ w,
                       const float* __restrict__ b, float* __restrict__ out) {
  int i = blockIdx.x * blockDim.x + threadIdx.x;
  if (i >= NBATCH * 500) return;
  int o = i % 500, bb = i / 500;
  const float* xr = x + (size_t)bb * 10950;
  const float* wr = w + (size_t)o * 10950;
  float s = b[o];
  for (int k = 0; k < 10950; ++k) s += xr[k] * wr[k];
  out[i] = fmaxf(s, 0.f);
}

__global__ void k_lin2(const float* __restrict__ x, const float* __restrict__ w,
                       const float* __restrict__ b, float* __restrict__ out) {
  int i = blockIdx.x * blockDim.x + threadIdx.x;
  if (i >= NBATCH * 4) return;
  int o = i % 4, bb = i / 4;
  const float* xr = x + (size_t)bb * 500;
  const float* wr = w + (size_t)o * 500;
  float s = b[o];
  for (int k = 0; k < 500; ++k) s += xr[k] * wr[k];
  out[i] = fmaxf(s, 0.f);
}

// ---------------- launch ----------------
extern "C" void kernel_launch(void* const* d_in, const int* in_sizes, int n_in,
                              void* d_out, int out_size, void* d_ws, size_t ws_size,
                              hipStream_t stream) {
  (void)in_sizes; (void)n_in; (void)out_size; (void)ws_size;
  const int*   tokens = (const int*)d_in[0];
  const int*   eindex = (const int*)d_in[1];
  const int*   batch  = (const int*)d_in[2];
  const float* table  = (const float*)d_in[3];
  const float* Wggc   = (const float*)d_in[4];
  const float* Wih    = (const float*)d_in[5];
  const float* Whh    = (const float*)d_in[6];
  const float* bih    = (const float*)d_in[7];
  const float* bhh    = (const float*)d_in[8];
  const float* c1w = (const float*)d_in[9];  const float* c1b = (const float*)d_in[10];
  const float* c2w = (const float*)d_in[11]; const float* c2b = (const float*)d_in[12];
  const float* c3w = (const float*)d_in[13]; const float* c3b = (const float*)d_in[14];
  const float* l1w = (const float*)d_in[15]; const float* l1b = (const float*)d_in[16];
  const float* l2w = (const float*)d_in[17]; const float* l2b = (const float*)d_in[18];
  float* out = (float*)d_out;

  char* base = (char*)d_ws;
  size_t off = 0;
  auto alloc = [&](size_t bytes) -> char* {
    char* p = base + off;
    off += (bytes + 255) & ~(size_t)255;
    return p;
  };
  const size_t NH = (size_t)N_NODES * HDIM;
  const size_t HH = (size_t)HDIM * HDIM;

  float*          h     = (float*)          alloc(NH * 4);
  unsigned short* hbf   = (unsigned short*) alloc(NH * 2);
  float*          mbuf  = (float*)          alloc(NH * 4);   // reused as inn buffer
  float*          agg   = (float*)          alloc(NH * 4);
  unsigned short* aggbf = (unsigned short*) alloc(NH * 2);
  float*          rsum  = (float*)          alloc(NH * 4);
  float*          zsum  = (float*)          alloc(NH * 4);
  float*          hnb   = (float*)          alloc(NH * 4);
  unsigned short* wggc  = (unsigned short*) alloc((size_t)LAYERS * HH * 2);
  unsigned short* wih   = (unsigned short*) alloc(3 * HH * 2);
  unsigned short* whh   = (unsigned short*) alloc(3 * HH * 2);
  float*          pooled= (float*)          alloc((size_t)NBATCH * HDIM * 4);
  float*          c1    = (float*)          alloc((size_t)NBATCH * 50 * C1L * 4);
  float*          c2    = (float*)          alloc((size_t)NBATCH * 100 * C2L * 4);
  float*          c3    = (float*)          alloc((size_t)NBATCH * 150 * C3L * 4);
  float*          l1    = (float*)          alloc((size_t)NBATCH * 500 * 4);

  const int TB = 256;
  auto grid1 = [&](size_t n) { return dim3((unsigned)((n + TB - 1) / TB)); };

  // weights -> bf16 (deterministic, every launch)
  k_f32_to_bf16<<<grid1(LAYERS * HH), TB, 0, stream>>>(Wggc, wggc, (int)(LAYERS * HH));
  k_f32_to_bf16<<<grid1(3 * HH), TB, 0, stream>>>(Wih, wih, (int)(3 * HH));
  k_f32_to_bf16<<<grid1(3 * HH), TB, 0, stream>>>(Whh, whh, (int)(3 * HH));

  // h = pad(embed(tokens))
  k_embed<<<grid1(NH), TB, 0, stream>>>(tokens, table, h, hbf);

  dim3 ggrid((HDIM + BN - 1) / BN, (N_NODES + BM - 1) / BM);
  for (int L = 0; L < LAYERS; ++L) {
    // m = h @ W_ggc[L]   (B is K x N)
    k_gemm_bf16<0><<<ggrid, 256, 0, stream>>>(hbf, HDIM, wggc + (size_t)L * HH, HDIM,
                                              mbuf, HDIM, nullptr, 0, N_NODES, HDIM, HDIM);
    // agg = segment_sum(m[src], dst)
    k_zero_f32<<<grid1(NH), TB, 0, stream>>>(agg, (int)NH);
    k_scatter_add<<<dim3(E_EDGES), 256, 0, stream>>>(eindex, mbuf, agg);
    k_f32_to_bf16<<<grid1(NH), TB, 0, stream>>>(agg, aggbf, (int)NH);

    // r gate: rsum = agg@Wih_r^T + b_ih_r + h@Whh_r^T + b_hh_r
    k_gemm_bf16<1><<<ggrid, 256, 0, stream>>>(aggbf, HDIM, wih + 0 * HH, HDIM,
                                              rsum, HDIM, bih + 0 * HDIM, 0, N_NODES, HDIM, HDIM);
    k_gemm_bf16<1><<<ggrid, 256, 0, stream>>>(hbf, HDIM, whh + 0 * HH, HDIM,
                                              rsum, HDIM, bhh + 0 * HDIM, 1, N_NODES, HDIM, HDIM);
    // z gate
    k_gemm_bf16<1><<<ggrid, 256, 0, stream>>>(aggbf, HDIM, wih + 1 * HH, HDIM,
                                              zsum, HDIM, bih + 1 * HDIM, 0, N_NODES, HDIM, HDIM);
    k_gemm_bf16<1><<<ggrid, 256, 0, stream>>>(hbf, HDIM, whh + 1 * HH, HDIM,
                                              zsum, HDIM, bhh + 1 * HDIM, 1, N_NODES, HDIM, HDIM);
    // n gate (kept separate: n = tanh(inn + r*hn))
    k_gemm_bf16<1><<<ggrid, 256, 0, stream>>>(aggbf, HDIM, wih + 2 * HH, HDIM,
                                              mbuf, HDIM, bih + 2 * HDIM, 0, N_NODES, HDIM, HDIM);
    k_gemm_bf16<1><<<ggrid, 256, 0, stream>>>(hbf, HDIM, whh + 2 * HH, HDIM,
                                              hnb, HDIM, bhh + 2 * HDIM, 0, N_NODES, HDIM, HDIM);
    // GRU update
    k_gru<<<grid1(NH), TB, 0, stream>>>(rsum, zsum, mbuf, hnb, h, hbf);
  }

  // pooled = segment_max(relu(h), batch)
  k_zero_f32<<<grid1((size_t)NBATCH * HDIM), TB, 0, stream>>>(pooled, NBATCH * HDIM);
  k_relu_segmax<<<dim3(N_NODES), 256, 0, stream>>>(h, batch, pooled);

  // conv stack + MLP head
  k_conv1<<<grid1((size_t)NBATCH * 50 * C1L), TB, 0, stream>>>(pooled, c1w, c1b, c1);
  k_conv2<<<grid1((size_t)NBATCH * 100 * C2L), TB, 0, stream>>>(c1, c2w, c2b, c2);
  k_conv3<<<grid1((size_t)NBATCH * 150 * C3L), TB, 0, stream>>>(c2, c3w, c3b, c3);
  k_lin1<<<grid1((size_t)NBATCH * 500), TB, 0, stream>>>(c3, l1w, l1b, l1);
  k_lin2<<<grid1((size_t)NBATCH * 4), TB, 0, stream>>>(l1, l2w, l2b, out);
}
